// LSTMClassifier_32169305047231
// MI455X (gfx1250) — matrix operand order repaired and checked by reading, not compiled, not run on hardware
//
#include <hip/hip_runtime.h>
#include <hip/hip_bf16.h>

// ---------------------------------------------------------------------------
// LSTM classifier forward for gfx1250 (MI455X), bf16 WMMA path.
//   B=64, T=512, V=32000, E=512, H=1024, C=4, gates G=4H=4096, K=E+H=1536
// Step kernel: one wave per (gate, 16x16 tile) => 1024 waves/step; gate tiles
// combined through LDS in a fused LSTM-cell epilogue.
// ---------------------------------------------------------------------------

#define B_  64
#define T_  512
#define E_  512
#define H_  1024
#define G_  4096      // 4*H
#define KC_ 1536      // E + H
#define C_  4

typedef __attribute__((ext_vector_type(16))) __bf16 v16bf;
typedef __attribute__((ext_vector_type(8)))  __bf16 v8bf;
typedef __attribute__((ext_vector_type(8)))  float  v8f;

// ---- one-time prep kernels -------------------------------------------------

// Wt[n][k] = (k<E ? Wi[k][n] : Wh[k-E][n]) as bf16, n in [0,4096), k in [0,1536)
__global__ void build_wt_kernel(const float* __restrict__ Wi,
                                const float* __restrict__ Wh,
                                __bf16* __restrict__ Wt) {
  int idx = blockIdx.x * blockDim.x + threadIdx.x;
  if (idx >= G_ * KC_) return;
  int n = idx / KC_;
  int k = idx % KC_;
  float v = (k < E_) ? Wi[(size_t)k * G_ + n] : Wh[(size_t)(k - E_) * G_ + n];
  Wt[idx] = (__bf16)v;
}

// xb laid out [T][B][E] bf16: per-timestep A matrices are contiguous row-major.
__global__ void embed_bf16_kernel(const int* __restrict__ tokens,
                                  const float* __restrict__ emb,
                                  __bf16* __restrict__ xb) {
  int idx = blockIdx.x * blockDim.x + threadIdx.x;  // over T*B*E
  if (idx >= T_ * B_ * E_) return;
  int e  = idx % E_;
  int tb = idx / E_;
  int b  = tb % B_;
  int t  = tb / B_;
  int tok = tokens[b * T_ + t];
  xb[idx] = (__bf16)emb[(size_t)tok * E_ + e];
}

__global__ void init_state_kernel(float* __restrict__ c, __bf16* __restrict__ hb0) {
  int idx = blockIdx.x * blockDim.x + threadIdx.x;
  if (idx >= B_ * H_) return;
  c[idx]   = 0.0f;
  hb0[idx] = (__bf16)0.0f;
}

// ---- fused recurrent step --------------------------------------------------

__device__ __forceinline__ float sigmoidf_(float x) { return 1.0f / (1.0f + __expf(-x)); }

// Block = 128 threads = 4 waves; wave g computes gate g's 16x16 tile for this
// (batch-tile, hidden-tile). Gate tiles meet in LDS for the cell update.
__global__ void __launch_bounds__(128)
lstm_step_kernel(const __bf16* __restrict__ xt,    // [B, E] bf16, this timestep
                 const __bf16* __restrict__ hin,   // [B, H] bf16, h_{t-1}
                 const __bf16* __restrict__ Wt,    // [G, KC] bf16 (transposed W)
                 const float*  __restrict__ bias,  // [G]
                 float*        __restrict__ c,     // [B, H] f32 (in/out)
                 float*        __restrict__ h,     // [B, H] f32 (out)
                 __bf16*       __restrict__ hout)  // [B, H] bf16 (out)
{
  __shared__ float gbuf[4][16 * 16];     // i,f,g,o tiles (4 KB)

  const int g    = threadIdx.x >> 5;     // gate index = wave id (0..3)
  const int lane = threadIdx.x & 31;
  const int nh   = blockIdx.x * 16;      // hidden-dim tile base
  const int mb   = blockIdx.y * 16;      // batch tile base
  const int lr   = lane & 15;
  const int hi   = lane >> 4;            // lane-half select

  const int col  = nh + lr;              // hidden column (B/C fragment column)
  const int arow = mb + lr;              // batch row (A fragment row)

  // Bias-initialized accumulator (C-layout: value depends only on column).
  v8f acc;
  {
    float bv = bias[g * H_ + col];
#pragma unroll
    for (int r = 0; r < 8; ++r) acc[r] = bv;
  }

  // Per-lane base pointers; A and B both use contiguous k = 16*hi + s order.
  const __bf16* wrow   = Wt + (size_t)(g * H_ + col) * KC_ + hi * 16;  // B: 16 contig bf16
  const __bf16* arow_x = xt  + (size_t)arow * E_ + hi * 16;            // A: 2x8 contig bf16
  const __bf16* arow_h = hin + (size_t)arow * H_ + hi * 16;

  // K segment 1: x_t contribution (K = 0..E)
#pragma unroll 4
  for (int kt = 0; kt < E_; kt += 32) {
    v8bf alo = *(const v8bf*)(arow_x + kt);        // K = k0 .. k0+7
    v8bf ahi = *(const v8bf*)(arow_x + kt + 8);    // K = k0+8 .. k0+15
    v16bf a = __builtin_shufflevector(alo, ahi,
                0, 1, 2, 3, 4, 5, 6, 7, 8, 9, 10, 11, 12, 13, 14, 15);
    __builtin_prefetch(wrow + kt + 256, 0, 1);     // weight stream (L2-resident)
    v16bf bm = *(const v16bf*)(wrow + kt);         // 32B contiguous
    acc = __builtin_amdgcn_wmma_f32_16x16x32_bf16(
        false, a, false, bm, (short)0, acc, false, false);
  }

  // K segment 2: h_{t-1} contribution (K = E..E+H)
#pragma unroll 4
  for (int kt = 0; kt < H_; kt += 32) {
    v8bf alo = *(const v8bf*)(arow_h + kt);
    v8bf ahi = *(const v8bf*)(arow_h + kt + 8);
    v16bf a = __builtin_shufflevector(alo, ahi,
                0, 1, 2, 3, 4, 5, 6, 7, 8, 9, 10, 11, 12, 13, 14, 15);
    __builtin_prefetch(wrow + E_ + kt + 256, 0, 1);
    v16bf bm = *(const v16bf*)(wrow + E_ + kt);
    acc = __builtin_amdgcn_wmma_f32_16x16x32_bf16(
        false, a, false, bm, (short)0, acc, false, false);
  }

  // Deposit gate tile in LDS. C/D layout: row-in-tile = hi*8 + r, col = lr.
#pragma unroll
  for (int r = 0; r < 8; ++r)
    gbuf[g][(hi * 8 + r) * 16 + lr] = acc[r];

  __syncthreads();

  // Cooperative fused epilogue: 128 threads cover the 256-element tile.
#pragma unroll
  for (int e = threadIdx.x; e < 256; e += 128) {
    int m_in   = e >> 4;
    int col_in = e & 15;
    size_t idx = (size_t)(mb + m_in) * H_ + (nh + col_in);
    float iv = sigmoidf_(gbuf[0][e]);
    float fv = sigmoidf_(gbuf[1][e]);
    float gv = tanhf(gbuf[2][e]);
    float ov = sigmoidf_(gbuf[3][e]);
    float cv = fv * c[idx] + iv * gv;
    c[idx] = cv;
    float hv = ov * tanhf(cv);
    h[idx] = hv;
    hout[idx] = (__bf16)hv;
  }
}

// ---- classifier head -------------------------------------------------------

__global__ void head_kernel(const float* __restrict__ h,
                            const float* __restrict__ Wd,
                            const float* __restrict__ bd,
                            float* __restrict__ out) {
  int idx = threadIdx.x;            // 256 = B*C outputs
  int b  = idx >> 2;
  int cc = idx & 3;
  float s = bd[cc];
  const float* hr = h + (size_t)b * H_;
  for (int k = 0; k < H_; ++k) s += hr[k] * Wd[k * C_ + cc];
  out[idx] = s;
}

// ---- launch ----------------------------------------------------------------

extern "C" void kernel_launch(void* const* d_in, const int* in_sizes, int n_in,
                              void* d_out, int out_size, void* d_ws, size_t ws_size,
                              hipStream_t stream) {
  const int*   tokens = (const int*)d_in[0];
  const float* emb    = (const float*)d_in[1];
  const float* Wi     = (const float*)d_in[2];
  const float* Wh     = (const float*)d_in[3];
  const float* bias   = (const float*)d_in[4];
  const float* Wd     = (const float*)d_in[5];
  const float* bd     = (const float*)d_in[6];
  // d_in[7] = train flag: 0 => dropout is identity; ignored.

  // Workspace layout (~47 MB total), all offsets 32B-aligned.
  char* ws = (char*)d_ws;
  size_t o = 0;
  __bf16* Wt  = (__bf16*)(ws + o); o += (size_t)G_ * KC_ * sizeof(__bf16);      // 12.6 MB
  __bf16* xb  = (__bf16*)(ws + o); o += (size_t)T_ * B_ * E_ * sizeof(__bf16);  // 33.6 MB
  float*  c   = (float*)(ws + o);  o += (size_t)B_ * H_ * sizeof(float);
  float*  h   = (float*)(ws + o);  o += (size_t)B_ * H_ * sizeof(float);
  __bf16* hb0 = (__bf16*)(ws + o); o += (size_t)B_ * H_ * sizeof(__bf16);
  __bf16* hb1 = (__bf16*)(ws + o); o += (size_t)B_ * H_ * sizeof(__bf16);

  // One-time (per launch) parallel prep.
  hipLaunchKernelGGL(build_wt_kernel, dim3((G_ * KC_ + 255) / 256), dim3(256), 0, stream,
                     Wi, Wh, Wt);
  hipLaunchKernelGGL(embed_bf16_kernel, dim3((T_ * B_ * E_ + 255) / 256), dim3(256), 0, stream,
                     tokens, emb, xb);
  hipLaunchKernelGGL(init_state_kernel, dim3((B_ * H_ + 255) / 256), dim3(256), 0, stream,
                     c, hb0);

  // Serial recurrence: one fused WMMA kernel per timestep, h double-buffered.
  // 1024 waves/step: grid = (H/16 hidden tiles, B/16 batch tiles), 4 waves/block.
  __bf16* hp[2] = {hb0, hb1};
  for (int t = 0; t < T_; ++t) {
    const __bf16* xt = xb + (size_t)t * B_ * E_;
    hipLaunchKernelGGL(lstm_step_kernel, dim3(H_ / 16, B_ / 16), dim3(128), 0, stream,
                       xt, hp[t & 1], Wt, bias, c, h, hp[(t + 1) & 1]);
  }

  hipLaunchKernelGGL(head_kernel, dim3(1), dim3(256), 0, stream,
                     h, Wd, bd, (float*)d_out);
}
